// CrossBlock_60816736911779
// MI455X (gfx1250) — compile-verified, matrix operand order checked by reading, not run on hardware
//
#include <hip/hip_runtime.h>
#include <cstddef>
#include <cstdint>

typedef __attribute__((ext_vector_type(16))) _Float16 v16h;
typedef __attribute__((ext_vector_type(8)))  _Float16 v8h;
typedef __attribute__((ext_vector_type(8)))  float    v8f;

// ---------------------------------------------------------------- fragments
__device__ __forceinline__ int lane_id() { return (int)(threadIdx.x & 31); }

// Unified fragment load (two contiguous 16B loads per lane).
// A operand (16x32 MxK) from row-major s[m*ld + k], lane row = l&15.
// B operand (32x16 KxN) from N-major    s[n*ld + k], lane row = l&15.
// ld must be a multiple of 8 halves; s must be 16B aligned.
__device__ __forceinline__ v16h frag_ld(const _Float16* s, int ld) {
  const int l = lane_id();
  const _Float16* p = s + (l & 15) * ld + ((l >> 4) << 3);
  v8h lo = *(const v8h*)p;
  v8h hi = *(const v8h*)(p + 16);
  return __builtin_shufflevector(lo, hi, 0, 1, 2, 3, 4, 5, 6, 7,
                                 8, 9, 10, 11, 12, 13, 14, 15);
}

__device__ __forceinline__ v8f wmma_f16(v16h a, v16h b, v8f c) {
  return __builtin_amdgcn_wmma_f32_16x16x32_f16(false, a, false, b, (short)0, c,
                                                false, false);
}

// ---------------------------------------------------------------- TDM path
#if defined(__gfx1250__) && __has_builtin(__builtin_amdgcn_tensor_load_to_lds)
#define USE_TDM 1
#else
#define USE_TDM 0
#endif

#if USE_TDM
typedef unsigned int u32x4 __attribute__((ext_vector_type(4)));
typedef int          i32x4 __attribute__((ext_vector_type(4)));
typedef int          i32x8 __attribute__((ext_vector_type(8)));

// 2-D tensor DMA: tile_h rows x tile_w f16 elems, global row stride src_stride
// elems, LDS rows padded 64B data + 16B pad (matches AS_LD=40 halves).
// D# bit layout per CDNA5 ISA 8.3/8.4.
__device__ __forceinline__ void tdm_load_2d(const _Float16* gsrc, _Float16* ldst,
                                            unsigned tile_w, unsigned tile_h,
                                            unsigned src_stride) {
  const unsigned long long ga = (unsigned long long)(uintptr_t)gsrc;
  const unsigned la = (unsigned)(uintptr_t)ldst;  // low 32 bits = LDS byte offset
  u32x4 g0;
  g0[0] = 1u;                                  // count=1 (valid user descriptor)
  g0[1] = la;                                  // lds_addr (bytes)
  g0[2] = (unsigned)ga;                        // global_addr[31:0]
  g0[3] = (unsigned)((ga >> 32) & 0x1FFFFFFu) | (2u << 30);  // addr[56:32]|type=2
  i32x8 g1;
  // data_size=1 (2B) | pad_enable | pad_interval=3 (64B) | pad_amount=3 (16B)
  g1[0] = (int)((1u << 16) | (1u << 20) | (3u << 22) | (3u << 25));
  g1[1] = (int)((src_stride & 0xFFFFu) << 16);                  // tensor_dim0 lo16
  g1[2] = (int)(((src_stride >> 16) & 0xFFFFu) |
                ((tile_h & 0xFFFFu) << 16));                    // dim0 hi | dim1 lo
  g1[3] = (int)(((tile_h >> 16) & 0xFFFFu) |
                ((tile_w & 0xFFFFu) << 16));                    // dim1 hi | tile_dim0
  g1[4] = (int)(tile_h & 0xFFFFu);                              // tile_dim1 (dim2=0)
  g1[5] = (int)src_stride;                                      // dim0_stride[31:0]
  g1[6] = 0;                                                    // stride hi bits
  g1[7] = 0;
  i32x4 z4 = {};
#if __has_include(<hip/amd_detail/amd_gfx1250_TDM.h>)
  i32x8 z8 = {};
  __builtin_amdgcn_tensor_load_to_lds(g0, g1, z4, z4, z8, 0);   // clang-23 form
#else
  __builtin_amdgcn_tensor_load_to_lds(g0, g1, z4, z4, 0);       // ROCm 7.2 form
#endif
}
#define TDM_FENCE()                                           \
  do {                                                        \
    if ((threadIdx.x >> 5) == 0) __builtin_amdgcn_s_wait_tensorcnt(0); \
  } while (0)
#else
#define TDM_FENCE() do { } while (0)
#endif

// ---------------------------------------------------------------- converters
__global__ void k_cvt(const float* __restrict__ s, _Float16* __restrict__ d, int n) {
  int i = blockIdx.x * blockDim.x + threadIdx.x;
  if (i < n) d[i] = (_Float16)s[i];
}

// ---------------------------------------------------------------- axial MHA
// One workgroup = one (batch, row). 8 waves, 256 threads.
// LDS layout (halves, padded strides to spread banks):
//   Xs [64 x 136]  pos-major X slice            @ 0       (17408 B)
//   Qt [64 x 520]  pos-major Q (reused as F)    @ 8704    (66560 B)
//   Kt [64 x 520]  pos-major K                  @ 41984   (66560 B)
//   Vs [512 x 72]  chan-major V                 @ 75264   (73728 B)
//   Ps [4 x 64 x 64] probs f16                  @ 112128  (32768 B)
//   Ss [4 x 64 x 66] scores f32                 @ 128512  (67584 B)
// total = 324608 bytes
#define XS_LD 136
#define QK_LD 520
#define VS_LD 72
#define SS_LD 66
#define SMEM_BYTES 324608

template <bool POSMAJOR>
__device__ __forceinline__ void qkv_one(const _Float16* __restrict__ W,
                                        const float* __restrict__ bias,
                                        const _Float16* Xs, _Float16* O, int ldo,
                                        int wv, int l) {
  const int row0 = (l >> 4) << 3;
  for (int mi = 0; mi < 4; ++mi) {
    const int m0 = (wv * 4 + mi) * 16;
    v8f acc[4] = {};
#pragma unroll
    for (int kk = 0; kk < 4; ++kk) {
      v16h a = frag_ld(W + (size_t)m0 * 128 + kk * 32, 128);  // reused over nt
#pragma unroll
      for (int nt = 0; nt < 4; ++nt) {
        v16h bf = frag_ld(Xs + (nt * 16) * XS_LD + kk * 32, XS_LD);
        acc[nt] = wmma_f16(a, bf, acc[nt]);
      }
    }
#pragma unroll
    for (int nt = 0; nt < 4; ++nt) {
      const int col = nt * 16 + (l & 15);
      if (POSMAJOR) {
        // O[pos][chan]: 8 accum values are channel-contiguous -> b128 store
        v8h hv;
#pragma unroll
        for (int rr = 0; rr < 8; ++rr)
          hv[rr] = (_Float16)(acc[nt][rr] + bias[m0 + row0 + rr]);
        *(v8h*)(O + col * ldo + m0 + row0) = hv;
      } else {
#pragma unroll
        for (int rr = 0; rr < 8; ++rr) {
          const int m = m0 + row0 + rr;
          O[m * ldo + col] = (_Float16)(acc[nt][rr] + bias[m]);
        }
      }
    }
  }
}

__global__ void __launch_bounds__(256)
k_axial_mha(const _Float16* __restrict__ x16, const float* __restrict__ xf,
            const _Float16* __restrict__ wq, const float* __restrict__ bq,
            const _Float16* __restrict__ wk, const float* __restrict__ bk,
            const _Float16* __restrict__ wv_, const float* __restrict__ bv,
            const _Float16* __restrict__ wp, const float* __restrict__ bp,
            const float* __restrict__ rel,
            _Float16* __restrict__ cat16,
            int cbase, int branch) {
  extern __shared__ _Float16 smem[];
  _Float16* Xs = smem;                    // [64][136]
  _Float16* Qt = smem + 8704;             // [64][520]  (later reused as F)
  _Float16* Kt = smem + 41984;            // [64][520]
  _Float16* Vs = smem + 75264;            // [512][72]
  _Float16* Ps = smem + 112128;           // [4][64][64]
  float*    Ss = (float*)(smem + 128512); // [4][64][66]

  const int tid = threadIdx.x;
  const int wv = tid >> 5;
  const int l = tid & 31;
  const int row0 = (l >> 4) << 3;
  const int b = blockIdx.x / 56;
  const int r = blockIdx.x % 56;

  const int pmul = branch ? 56 : 1;
  const int poff = branch ? r : r * 56;

  // ---- stage X slice pos-major: Xs[j][c], rows 56..63 zeroed ----
  const _Float16* xb = x16 + ((size_t)b * 256 + cbase) * 3136;
  for (int idx = tid; idx < 64 * 128; idx += 256) {
    const int j = idx >> 7, c = idx & 127;
    Xs[j * XS_LD + c] = (j < 56) ? xb[c * 3136 + j * pmul + poff] : (_Float16)0.f;
  }
  __syncthreads();

  // ---- QKV projections: [512x128] x [128x64] ----
  qkv_one<true>(wq, bq, Xs, Qt, QK_LD, wv, l);
  qkv_one<true>(wk, bk, Xs, Kt, QK_LD, wv, l);
  qkv_one<false>(wv_, bv, Xs, Vs, VS_LD, wv, l);
  __syncthreads();

  // ---- S = Q K^T * scale + rel bias (Q,K pos-major -> both frag_ld) ----
  {
    const float scale = 0.088388347648318447f; // 128^-0.5
    const int head = wv >> 1;
    const int itb = (wv & 1) * 2;
    for (int it = itb; it < itb + 2; ++it) {
      v8f acc[4] = {};
#pragma unroll
      for (int kk = 0; kk < 4; ++kk) {
        v16h a = frag_ld(Qt + (it * 16) * QK_LD + head * 128 + kk * 32, QK_LD);
#pragma unroll
        for (int jt = 0; jt < 4; ++jt) {
          v16h bf = frag_ld(Kt + (jt * 16) * QK_LD + head * 128 + kk * 32, QK_LD);
          acc[jt] = wmma_f16(a, bf, acc[jt]);
        }
      }
#pragma unroll
      for (int jt = 0; jt < 4; ++jt) {
        const int j = jt * 16 + (l & 15);
#pragma unroll
        for (int rr = 0; rr < 8; ++rr) {
          const int i = it * 16 + rr + row0;
          float s = -1e30f;
          if (i < 56 && j < 56)
            s = acc[jt][rr] * scale + rel[head * 3136 + (i - j + 55)];
          Ss[(head * 64 + i) * SS_LD + j] = s;
        }
      }
    }
  }
  __syncthreads();

  // ---- softmax rows -> Ps (f16, row-major [i][j], padded with zeros) ----
  if (tid < 224) {
    const int head = tid / 56, i = tid % 56;
    float* srow = Ss + (head * 64 + i) * SS_LD;
    _Float16* prow = Ps + (head * 64 + i) * 64;
    float mx = -1e30f;
    for (int j = 0; j < 56; ++j) mx = fmaxf(mx, srow[j]);
    float sum = 0.f;
    for (int j = 0; j < 56; ++j) {
      const float e = __expf(srow[j] - mx);
      sum += e;
      prow[j] = (_Float16)e;
    }
    const float inv = 1.f / sum;
    for (int j = 0; j < 56; ++j) prow[j] = (_Float16)((float)prow[j] * inv);
    for (int j = 56; j < 64; ++j) prow[j] = (_Float16)0.f;
  } else {
    const int u = tid - 224;  // zero the 8 pad rows of each head
    const int head = u >> 3, i = 56 + (u & 7);
    _Float16* prow = Ps + (head * 64 + i) * 64;
    for (int j = 0; j < 64; ++j) prow[j] = (_Float16)0.f;
  }
  __syncthreads();

  // ---- O = P @ V^T -> flat F (torch raw-reshape layout), reuses Qt ----
  // f = head*7168 + i*128 + d ; stored at F[(f>>9)*520 + (f&511)]
  {
    const int head = wv >> 1;
    const int dtb = (wv & 1) * 4;
    for (int it = 0; it < 4; ++it) {
      v8f acc[4] = {};
#pragma unroll
      for (int kk = 0; kk < 2; ++kk) {
        v16h a = frag_ld(Ps + (head * 64 + it * 16) * 64 + kk * 32, 64);
#pragma unroll
        for (int dq = 0; dq < 4; ++dq) {
          v16h bf = frag_ld(Vs + (head * 128 + (dtb + dq) * 16) * VS_LD + kk * 32,
                            VS_LD);
          acc[dq] = wmma_f16(a, bf, acc[dq]);
        }
      }
#pragma unroll
      for (int dq = 0; dq < 4; ++dq) {
        const int d = (dtb + dq) * 16 + (l & 15);
#pragma unroll
        for (int rr = 0; rr < 8; ++rr) {
          const int i = it * 16 + rr + row0;
          if (i < 56) {
            const int f = head * 7168 + i * 128 + d;
            Qt[(f >> 9) * QK_LD + (f & 511)] = (_Float16)acc[dq][rr];
          }
        }
      }
    }
  }
  __syncthreads();

  // ---- out projection + bias + residual -> cat16 ----
  {
    const _Float16* F = Qt;
    const int m0 = wv * 16;  // 8 waves -> 128 output channels
    v8f acc[4] = {};
    for (int kk = 0; kk < 16; ++kk) {
      v16h a = frag_ld(wp + (size_t)m0 * 512 + kk * 32, 512);  // reused over nt
#pragma unroll
      for (int nt = 0; nt < 4; ++nt) {
        v16h bf = frag_ld(F + (nt * 16) * QK_LD + kk * 32, QK_LD);
        acc[nt] = wmma_f16(a, bf, acc[nt]);
      }
    }
#pragma unroll
    for (int nt = 0; nt < 4; ++nt) {
      const int wcol = nt * 16 + (l & 15);
      if (wcol < 56) {
        const int pix = wcol * pmul + poff;
#pragma unroll
        for (int rr = 0; rr < 8; ++rr) {
          const int c = m0 + row0 + rr;
          const size_t o = ((size_t)b * 256 + cbase + c) * 3136 + pix;
          cat16[o] = (_Float16)(acc[nt][rr] + bp[c] + xf[o]);
        }
      }
    }
  }
}

// ---------------------------------------------------------------- GEMMs
// Block tile: 128 (M) x 64 (N), 8 waves; wave computes a 16x64 strip.
// Double-buffered: As [2][128][40] row-major, Bs [2][64][40] N-major.
// A tile staged by the Tensor Data Mover when available (one TDM op per
// k-step, LDS row padding 64B+16B done by D# pad fields), else per-lane b128.
#define AS_LD 40
#define BS_LD 40
#define AS_TILE (128 * AS_LD)
#define BS_TILE (64 * BS_LD)

template <int K>
__device__ __forceinline__ void gemm_tile(const _Float16* __restrict__ W,
                                          const _Float16* __restrict__ act,
                                          v8f acc[4], _Float16* As, _Float16* Bs,
                                          int m0, int p0) {
  const int tid = threadIdx.x;
  const int wvv = tid >> 5;
#pragma unroll
  for (int nt = 0; nt < 4; ++nt) acc[nt] = (v8f){};

  auto stage = [&](int kk, int buf) {
    _Float16* dA = As + buf * AS_TILE;
    _Float16* dB = Bs + buf * BS_TILE;
#if USE_TDM
    if (tid == 0)  // one TDM per workgroup; EXEC ignored by tensor ops
      tdm_load_2d(W + (size_t)m0 * K + kk, dA, /*w*/ 32, /*h*/ 128, /*stride*/ K);
#else
    {
      const int mrow = tid >> 1;
      const int k16 = (tid & 1) * 16;
      const _Float16* src = W + (size_t)(m0 + mrow) * K + kk + k16;
      *(v8h*)(dA + mrow * AS_LD + k16) = *(const v8h*)src;
      *(v8h*)(dA + mrow * AS_LD + k16 + 8) = *(const v8h*)(src + 8);
    }
#endif
    // B transposed: vector global read along pixels, scatter to [n][k]
    const int krow = tid >> 3;
    const int n8 = (tid & 7) * 8;
    v8h row = *(const v8h*)(act + (size_t)(kk + krow) * 3136 + p0 + n8);
#pragma unroll
    for (int i = 0; i < 8; ++i) dB[(n8 + i) * BS_LD + krow] = row[i];
  };

  stage(0, 0);
  TDM_FENCE();
  __syncthreads();

  int p = 0;
  for (int kk = 0; kk < K; kk += 32) {
    if (kk + 32 < K) stage(kk + 32, p ^ 1);  // overlap copy with compute
    const _Float16* cA = As + p * AS_TILE;
    const _Float16* cB = Bs + p * BS_TILE;
    v16h a = frag_ld(cA + wvv * 16 * AS_LD, AS_LD);
#pragma unroll
    for (int nt = 0; nt < 4; ++nt) {
      v16h bf = frag_ld(cB + nt * 16 * BS_LD, BS_LD);
      acc[nt] = wmma_f16(a, bf, acc[nt]);
    }
    TDM_FENCE();
    __syncthreads();
    p ^= 1;
  }
}

__global__ void __launch_bounds__(256)
k_gemm_cp(const _Float16* __restrict__ w, const float* __restrict__ bias,
          const _Float16* __restrict__ act, const float* __restrict__ ls1,
          const float* __restrict__ xf, float* __restrict__ net,
          _Float16* __restrict__ net16) {
  __shared__ __align__(16) _Float16 As[2 * AS_TILE];
  __shared__ __align__(16) _Float16 Bs[2 * BS_TILE];
  const int p0 = blockIdx.x * 64, m0 = blockIdx.y * 128, b = blockIdx.z;
  v8f acc[4];
  gemm_tile<256>(w, act + (size_t)b * 256 * 3136, acc, As, Bs, m0, p0);
  const int l = threadIdx.x & 31, wv = threadIdx.x >> 5;
  const int row0 = (l >> 4) << 3;
#pragma unroll
  for (int nt = 0; nt < 4; ++nt) {
    const int p = p0 + nt * 16 + (l & 15);
#pragma unroll
    for (int rr = 0; rr < 8; ++rr) {
      const int c = m0 + wv * 16 + rr + row0;
      const size_t o = ((size_t)b * 256 + c) * 3136 + p;
      float v = acc[nt][rr] + bias[c];
      v = v * ls1[c] + xf[o];
      net[o] = v;
      net16[o] = (_Float16)v;
    }
  }
}

__global__ void __launch_bounds__(256)
k_gemm_m1(const _Float16* __restrict__ w, const float* __restrict__ bias,
          const _Float16* __restrict__ act, _Float16* __restrict__ h16) {
  __shared__ __align__(16) _Float16 As[2 * AS_TILE];
  __shared__ __align__(16) _Float16 Bs[2 * BS_TILE];
  const int p0 = blockIdx.x * 64, m0 = blockIdx.y * 128, b = blockIdx.z;
  v8f acc[4];
  gemm_tile<256>(w, act + (size_t)b * 256 * 3136, acc, As, Bs, m0, p0);
  const int l = threadIdx.x & 31, wv = threadIdx.x >> 5;
  const int row0 = (l >> 4) << 3;
#pragma unroll
  for (int nt = 0; nt < 4; ++nt) {
    const int p = p0 + nt * 16 + (l & 15);
#pragma unroll
    for (int rr = 0; rr < 8; ++rr) {
      const int c = m0 + wv * 16 + rr + row0;
      float v = acc[nt][rr] + bias[c];
      v = 0.5f * v * (1.f + erff(v * 0.70710678118654752f));  // exact GELU
      h16[((size_t)b * 512 + c) * 3136 + p] = (_Float16)v;
    }
  }
}

__global__ void __launch_bounds__(256)
k_gemm_m2(const _Float16* __restrict__ w, const float* __restrict__ bias,
          const _Float16* __restrict__ act, const float* __restrict__ ls2,
          const float* __restrict__ net, float* __restrict__ out) {
  __shared__ __align__(16) _Float16 As[2 * AS_TILE];
  __shared__ __align__(16) _Float16 Bs[2 * BS_TILE];
  const int p0 = blockIdx.x * 64, m0 = blockIdx.y * 128, b = blockIdx.z;
  v8f acc[4];
  gemm_tile<512>(w, act + (size_t)b * 512 * 3136, acc, As, Bs, m0, p0);
  const int l = threadIdx.x & 31, wv = threadIdx.x >> 5;
  const int row0 = (l >> 4) << 3;
#pragma unroll
  for (int nt = 0; nt < 4; ++nt) {
    const int p = p0 + nt * 16 + (l & 15);
#pragma unroll
    for (int rr = 0; rr < 8; ++rr) {
      const int c = m0 + wv * 16 + rr + row0;
      const size_t o = ((size_t)b * 256 + c) * 3136 + p;
      out[o] = (acc[nt][rr] + bias[c]) * ls2[c] + net[o];
    }
  }
}

// ---------------------------------------------------------------- launch
extern "C" void kernel_launch(void* const* d_in, const int* in_sizes, int n_in,
                              void* d_out, int out_size, void* d_ws, size_t ws_size,
                              hipStream_t stream) {
  (void)in_sizes; (void)n_in; (void)out_size; (void)ws_size;
  const float* x    = (const float*)d_in[0];
  const float* xq_w = (const float*)d_in[1];  const float* xq_b = (const float*)d_in[2];
  const float* xk_w = (const float*)d_in[3];  const float* xk_b = (const float*)d_in[4];
  const float* xv_w = (const float*)d_in[5];  const float* xv_b = (const float*)d_in[6];
  const float* xp_w = (const float*)d_in[7];  const float* xp_b = (const float*)d_in[8];
  const float* rel_x = (const float*)d_in[9];
  const float* yq_w = (const float*)d_in[10]; const float* yq_b = (const float*)d_in[11];
  const float* yk_w = (const float*)d_in[12]; const float* yk_b = (const float*)d_in[13];
  const float* yv_w = (const float*)d_in[14]; const float* yv_b = (const float*)d_in[15];
  const float* yp_w = (const float*)d_in[16]; const float* yp_b = (const float*)d_in[17];
  const float* rel_y = (const float*)d_in[18];
  const float* cp_w = (const float*)d_in[19]; const float* cp_b = (const float*)d_in[20];
  const float* m1_w = (const float*)d_in[21]; const float* m1_b = (const float*)d_in[22];
  const float* m2_w = (const float*)d_in[23]; const float* m2_b = (const float*)d_in[24];
  const float* ls1  = (const float*)d_in[25]; const float* ls2  = (const float*)d_in[26];

  size_t off = 0;
  auto a16 = [&](size_t n) {
    _Float16* p = (_Float16*)((char*)d_ws + off);
    off = (off + n * 2 + 255) & ~(size_t)255;
    return p;
  };
  auto a32 = [&](size_t n) {
    float* p = (float*)((char*)d_ws + off);
    off = (off + n * 4 + 255) & ~(size_t)255;
    return p;
  };
  const size_t NPIX = (size_t)8 * 256 * 3136;
  _Float16* x16   = a16(NPIX);
  _Float16* cat16 = a16(NPIX);
  _Float16* net16 = a16(NPIX);
  _Float16* h16   = a16((size_t)8 * 512 * 3136);
  float*    net   = a32(NPIX);
  _Float16* xq16 = a16(512 * 128); _Float16* xk16 = a16(512 * 128);
  _Float16* xv16 = a16(512 * 128); _Float16* xp16 = a16(128 * 512);
  _Float16* yq16 = a16(512 * 128); _Float16* yk16 = a16(512 * 128);
  _Float16* yv16 = a16(512 * 128); _Float16* yp16 = a16(128 * 512);
  _Float16* cp16 = a16(256 * 256);
  _Float16* m116 = a16(512 * 256);
  _Float16* m216 = a16(256 * 512);

  auto cvt = [&](const float* s, _Float16* d, int n) {
    k_cvt<<<(n + 255) / 256, 256, 0, stream>>>(s, d, n);
  };
  cvt(x, x16, (int)NPIX);
  cvt(xq_w, xq16, 65536); cvt(xk_w, xk16, 65536); cvt(xv_w, xv16, 65536);
  cvt(xp_w, xp16, 65536);
  cvt(yq_w, yq16, 65536); cvt(yk_w, yk16, 65536); cvt(yv_w, yv16, 65536);
  cvt(yp_w, yp16, 65536);
  cvt(cp_w, cp16, 65536); cvt(m1_w, m116, 131072); cvt(m2_w, m216, 131072);

  const size_t smem = SMEM_BYTES;  // 324608 B dynamic LDS (<= 320 KB WGP)
  (void)hipFuncSetAttribute(reinterpret_cast<const void*>(k_axial_mha),
                            hipFuncAttributeMaxDynamicSharedMemorySize, (int)smem);
  k_axial_mha<<<448, 256, smem, stream>>>(x16, x, xq16, xq_b, xk16, xk_b, xv16,
                                          xv_b, xp16, xp_b, rel_x, cat16, 0, 0);
  k_axial_mha<<<448, 256, smem, stream>>>(x16, x, yq16, yq_b, yk16, yk_b, yv16,
                                          yv_b, yp16, yp_b, rel_y, cat16, 128, 1);

  dim3 gcp(49, 2, 8), gm1(49, 4, 8), gm2(49, 2, 8);
  k_gemm_cp<<<gcp, 256, 0, stream>>>(cp16, cp_b, cat16, ls1, x, net, net16);
  k_gemm_m1<<<gm1, 256, 0, stream>>>(m116, m1_b, net16, h16);
  k_gemm_m2<<<gm2, 256, 0, stream>>>(m216, m2_b, h16, ls2, net, (float*)d_out);
}
